// SelfAttention_78666620993755
// MI455X (gfx1250) — matrix operand order repaired and checked by reading, not compiled, not run on hardware
//
#include <hip/hip_runtime.h>
#include <hip/hip_bf16.h>
#include <math.h>

// ---------------- problem constants ----------------
#define B   2
#define S   2048
#define DM  4096
#define H   32
#define KV  8
#define HD  128
#define NREP (H / KV)

typedef __attribute__((ext_vector_type(16))) __bf16 v16bf;
typedef __attribute__((ext_vector_type(8)))  float  v8f;
typedef __attribute__((ext_vector_type(8)))  __bf16 bfx8;

union V16 { v16bf v; bfx8 h[2]; };

__device__ __forceinline__ v8f wmma_bf16(v16bf a, v16bf b, v8f c) {
  // D = A(16xK=32) * B(32x16) + C  (bf16 in, fp32 accum)
  return __builtin_amdgcn_wmma_f32_16x16x32_bf16(
      false, a, false, b, (short)0, c, false, false);
}

// CDNA5 async global->LDS copy (ASYNCcnt tracked). lds_byte is the byte offset
// within the workgroup's LDS (HW adds LDS_BASE); g is the per-lane global addr.
__device__ __forceinline__ void async_b128(unsigned lds_byte, const void* g) {
  asm volatile("global_load_async_to_lds_b128 %0, %1, off"
               :
               : "v"(lds_byte), "v"(g)
               : "memory");
}
__device__ __forceinline__ void wait_async0() {
  asm volatile("s_wait_asynccnt 0x0" ::: "memory");
}

// ---------------- elementwise fp32 -> bf16 ----------------
__global__ void cvt_bf16_kernel(const float* __restrict__ in,
                                __bf16* __restrict__ out, size_t n) {
  size_t i = (size_t)blockIdx.x * blockDim.x + threadIdx.x;
  if (i < n) out[i] = (__bf16)in[i];
}

// ---------------- transpose-convert: w[K,N] fp32 -> wt[N,K] bf16 ----------------
__global__ void wtrans_kernel(const float* __restrict__ w,
                              __bf16* __restrict__ wt, int K, int N) {
  __shared__ float t[32][33];
  int n0 = blockIdx.x * 32, k0 = blockIdx.y * 32;
  int tx = threadIdx.x, ty = threadIdx.y;  // (32, 8)
  for (int i = ty; i < 32; i += 8)
    t[i][tx] = w[(size_t)(k0 + i) * N + n0 + tx];
  __syncthreads();
  for (int i = ty; i < 32; i += 8)
    wt[(size_t)(n0 + i) * K + k0 + tx] = (__bf16)t[tx][i];
}

// ---------------- v[B,S,KV,HD] fp32 -> vT[B,KV,HD,S] bf16 ----------------
__global__ void vtrans_kernel(const float* __restrict__ vf,
                              __bf16* __restrict__ vT) {
  __shared__ float t[32][33];
  int bk = blockIdx.z;
  int b = bk / KV, kv = bk % KV;
  int hd0 = blockIdx.x * 32, s0 = blockIdx.y * 32;
  int tx = threadIdx.x, ty = threadIdx.y;  // (32, 8)
  for (int i = ty; i < 32; i += 8)
    t[i][tx] = vf[((size_t)(b * S + s0 + i) * KV + kv) * HD + hd0 + tx];
  __syncthreads();
  for (int i = ty; i < 32; i += 8)
    vT[((size_t)(b * KV + kv) * HD + hd0 + i) * S + s0 + tx] = (__bf16)t[tx][i];
}

// ---------------- RoPE + convert to bf16 (optional scale) ----------------
// xf layout [B,S,nh,HD], pairs (2p, 2p+1); cos/sin [S, HD/2]
__global__ void rope_kernel(const float* __restrict__ xf,
                            const float* __restrict__ cosT,
                            const float* __restrict__ sinT,
                            __bf16* __restrict__ out, int nh, float scale) {
  size_t idx = (size_t)blockIdx.x * blockDim.x + threadIdx.x;
  size_t total = (size_t)B * S * nh * (HD / 2);
  if (idx >= total) return;
  int p = (int)(idx & 63);
  size_t t = idx >> 6;          // (b,s,h)
  t /= (size_t)nh;              // (b,s)
  int s = (int)(t % S);
  size_t base = idx * 2;
  float xr = xf[base], xi = xf[base + 1];
  float c = cosT[s * (HD / 2) + p], sn = sinT[s * (HD / 2) + p];
  out[base]     = (__bf16)((xr * c - xi * sn) * scale);
  out[base + 1] = (__bf16)((xr * sn + xi * c) * scale);
}

// ---------------- GEMM: C[M,N] fp32 = A[M,K] bf16 * Bt[N,K]^T bf16 ----------------
// Dynamic LDS layout (base offset 0, no static LDS in this kernel):
//   As[2][128*LDT] at byte 0,  Bs[2][128*LDT] at byte 2*TILE_BYTES
#define LDT 40                      // padded LDS row stride (conflict-free reads)
#define TILE_HALFS (128 * LDT)      // 5120 bf16 per buffer
#define TILE_BYTES (TILE_HALFS * 2) // 10240 bytes
#define BS_BYTE_OFF (2 * TILE_BYTES)
#define GEMM_LDS_BYTES (4 * TILE_BYTES)  // 40960

__global__ __launch_bounds__(256) void gemm_bf16_kernel(
    const __bf16* __restrict__ A, const __bf16* __restrict__ Bt,
    float* __restrict__ C, int M, int N, int K) {
  extern __shared__ char smem_raw[];

  const int tid = threadIdx.x;
  const int lane = tid & 31, w = tid >> 5;
  const int wm = w & 3, wn = w >> 2;       // 4 x 2 wave grid
  const int lr = lane & 15, hi = lane >> 4;
  const int m0 = blockIdx.y * 128, n0 = blockIdx.x * 128;

  const int ldr = tid >> 1;                // 0..127 (tile row)
  const int ldc = (tid & 1) * 16;          // 0 or 16 (tile col)

  v8f zero = {0.f, 0.f, 0.f, 0.f, 0.f, 0.f, 0.f, 0.f};
  v8f acc[2][4];
#pragma unroll
  for (int mi = 0; mi < 2; ++mi)
#pragma unroll
    for (int ni = 0; ni < 4; ++ni) acc[mi][ni] = zero;

  // Async global->LDS staging of one 128x32 A tile + 128x32 Bt tile.
  auto load_tile = [&](int buf, int kk) {
    const __bf16* ap = A + (size_t)(m0 + ldr) * K + kk + ldc;
    const __bf16* bp = Bt + (size_t)(n0 + ldr) * K + kk + ldc;
    unsigned la = (unsigned)(buf * TILE_BYTES + (ldr * LDT + ldc) * 2);
    unsigned lb = la + (unsigned)BS_BYTE_OFF;
    async_b128(la, ap);
    async_b128(la + 16u, ap + 8);
    async_b128(lb, bp);
    async_b128(lb + 16u, bp + 8);
  };

  const int nk = K / 32;
  load_tile(0, 0);
  wait_async0();
  __syncthreads();

  for (int kt = 0; kt < nk; ++kt) {
    const int cur = kt & 1;
    if (kt + 1 < nk) load_tile(cur ^ 1, (kt + 1) * 32);
    if (kt + 2 < nk) {
      __builtin_prefetch(A + (size_t)(m0 + ldr) * K + (kt + 2) * 32 + ldc, 0, 0);
      __builtin_prefetch(Bt + (size_t)(n0 + ldr) * K + (kt + 2) * 32 + ldc, 0, 0);
    }
    const __bf16* as = (const __bf16*)(smem_raw + cur * TILE_BYTES);
    const __bf16* bs = (const __bf16*)(smem_raw + BS_BYTE_OFF + cur * TILE_BYTES);

    V16 a[2], bb[4];
#pragma unroll
    for (int mi = 0; mi < 2; ++mi) {
      int row = wm * 32 + mi * 16 + lr;   // A frag: lane=row, interleaved K halves
      a[mi].h[0] = *(const bfx8*)&as[row * LDT + hi * 8];
      a[mi].h[1] = *(const bfx8*)&as[row * LDT + 16 + hi * 8];
    }
#pragma unroll
    for (int ni = 0; ni < 4; ++ni) {
      int col = wn * 64 + ni * 16 + lr;   // B frag: lane=col, interleaved K halves
      bb[ni].h[0] = *(const bfx8*)&bs[col * LDT + hi * 8];
      bb[ni].h[1] = *(const bfx8*)&bs[col * LDT + 16 + hi * 8];
    }
#pragma unroll
    for (int mi = 0; mi < 2; ++mi)
#pragma unroll
      for (int ni = 0; ni < 4; ++ni)
        acc[mi][ni] = wmma_bf16(a[mi].v, bb[ni].v, acc[mi][ni]);

    wait_async0();     // next-tile async writes have landed in LDS
    __syncthreads();
  }

#pragma unroll
  for (int mi = 0; mi < 2; ++mi)
#pragma unroll
    for (int ni = 0; ni < 4; ++ni) {
      int brow = m0 + wm * 32 + mi * 16 + hi * 8;
      int col  = n0 + wn * 64 + ni * 16 + lr;
      float* cp = C + (size_t)brow * N + col;
#pragma unroll
      for (int i = 0; i < 8; ++i) cp[(size_t)i * N] = acc[mi][ni][i];
    }
}

// ---------------- flash attention (full softmax, GQA) ----------------
// q  [B,S,H,HD]  bf16 (RoPE'd, pre-scaled by 1/sqrt(HD))
// k  [B,S,KV,HD] bf16 (RoPE'd)
// vT [B,KV,HD,S] bf16
// ctx[B,S,H,HD]  bf16
__global__ __launch_bounds__(256) void attn_kernel(
    const __bf16* __restrict__ q, const __bf16* __restrict__ k,
    const __bf16* __restrict__ vT, __bf16* __restrict__ ctx) {
  const int lane = threadIdx.x & 31;
  const int wid = threadIdx.x >> 5;
  const int lr = lane & 15, hi = lane >> 4;
  const int qrow0 = blockIdx.x * 128 + wid * 16;
  const int h = blockIdx.y, b = blockIdx.z;
  const int kvh = h / NREP;

  // Preload Q^T B-fragments: lane = column q, interleaved K halves of head-dims.
  V16 qf[4];
  {
    const __bf16* qp = q + ((size_t)(b * S + qrow0 + lr) * H + h) * HD;
#pragma unroll
    for (int j = 0; j < 4; ++j) {
      qf[j].h[0] = *(const bfx8*)(qp + j * 32 + hi * 8);
      qf[j].h[1] = *(const bfx8*)(qp + j * 32 + hi * 8 + 16);
    }
  }

  v8f zero = {0.f, 0.f, 0.f, 0.f, 0.f, 0.f, 0.f, 0.f};
  v8f o[8];
#pragma unroll
  for (int c = 0; c < 8; ++c) o[c] = zero;
  float m_run = -3.0e38f, l_run = 0.f;

  for (int k0 = 0; k0 < S; k0 += 32) {
    // S^T tiles: D[k,q] = K_tile(16k x 32hd) x Q^T(32hd x 16q)
    v8f s0 = zero, s1 = zero;
    const __bf16* kp0 = k + ((size_t)(b * S + k0 + lr) * KV + kvh) * HD;
    const __bf16* kp1 = kp0 + (size_t)16 * KV * HD;
#pragma unroll
    for (int j = 0; j < 4; ++j) {
      int off = j * 32 + hi * 8;  // A frag: interleaved K halves
      V16 ka, kb1;
      ka.h[0]  = *(const bfx8*)(kp0 + off);
      ka.h[1]  = *(const bfx8*)(kp0 + off + 16);
      kb1.h[0] = *(const bfx8*)(kp1 + off);
      kb1.h[1] = *(const bfx8*)(kp1 + off + 16);
      s0 = wmma_bf16(ka.v, qf[j].v, s0);
      s1 = wmma_bf16(kb1.v, qf[j].v, s1);
    }

    // Online softmax over this 32-key chunk (row = q = lane%16).
    float cmax = -3.0e38f;
#pragma unroll
    for (int i = 0; i < 8; ++i) cmax = fmaxf(cmax, fmaxf(s0[i], s1[i]));
    cmax = fmaxf(cmax, __shfl_xor(cmax, 16));
    float mnew = fmaxf(m_run, cmax);
    float e0[8], e1[8], csum = 0.f;
#pragma unroll
    for (int i = 0; i < 8; ++i) {
      e0[i] = __expf(s0[i] - mnew);
      e1[i] = __expf(s1[i] - mnew);
      csum += e0[i] + e1[i];
    }
    csum += __shfl_xor(csum, 16);
    float alpha = __expf(m_run - mnew);
    l_run = l_run * alpha + csum;
    m_run = mnew;

    // Per-output-row rescale factors (row = hi*8 + i lives in lane (hi*8+i)).
    float f[8];
#pragma unroll
    for (int i = 0; i < 8; ++i) f[i] = __shfl(alpha, hi ? (i + 8) : i);

    // P A-fragment: lane = q row; halves line up with S^T C-layout exactly.
    V16 p;
#pragma unroll
    for (int i = 0; i < 8; ++i) {
      p.v[i]     = (__bf16)e0[i];
      p.v[i + 8] = (__bf16)e1[i];
    }

    // O += P(16q x 32k) x V(32k x 128hd), 8 column tiles of 16.
    const __bf16* vbase =
        vT + (size_t)(b * KV + kvh) * HD * S + k0 + hi * 8;
#pragma unroll
    for (int c = 0; c < 8; ++c) {
      const __bf16* vp = vbase + (size_t)(c * 16 + lr) * S;
      V16 vf_;
      vf_.h[0] = *(const bfx8*)(vp);
      vf_.h[1] = *(const bfx8*)(vp + 16);
      v8f oc = o[c];
#pragma unroll
      for (int i = 0; i < 8; ++i) oc[i] *= f[i];
      o[c] = wmma_bf16(p.v, vf_.v, oc);
    }
  }

  // Normalize and store ctx (bf16).
  float linv[8];
#pragma unroll
  for (int i = 0; i < 8; ++i)
    linv[i] = 1.0f / __shfl(l_run, hi ? (i + 8) : i);
#pragma unroll
  for (int c = 0; c < 8; ++c) {
    int col = c * 16 + lr;
#pragma unroll
    for (int i = 0; i < 8; ++i) {
      int row = qrow0 + hi * 8 + i;
      ctx[((size_t)(b * S + row) * H + h) * HD + col] =
          (__bf16)(o[c][i] * linv[i]);
    }
  }
}

// ---------------- host-side orchestration ----------------
extern "C" void kernel_launch(void* const* d_in, const int* in_sizes, int n_in,
                              void* d_out, int out_size, void* d_ws, size_t ws_size,
                              hipStream_t stream) {
  (void)in_sizes; (void)n_in; (void)out_size; (void)ws_size;
  const float* x    = (const float*)d_in[0];
  const float* wq   = (const float*)d_in[3];
  const float* wk   = (const float*)d_in[4];
  const float* wv   = (const float*)d_in[5];
  const float* wo   = (const float*)d_in[6];
  const float* fcos = (const float*)d_in[7];
  const float* fsin = (const float*)d_in[8];
  float* out = (float*)d_out;

  const size_t nBSD = (size_t)B * S * DM;        // 16777216
  const size_t nKVp = (size_t)B * S * KV * HD;   //  4194304

  char* ws = (char*)d_ws;
  size_t off = 0;
  auto alloc = [&](size_t bytes) { char* p = ws + off; off += (bytes + 255) & ~(size_t)255; return p; };
  __bf16* xb   = (__bf16*)alloc(nBSD * 2);
  __bf16* wqT  = (__bf16*)alloc((size_t)DM * DM * 2);
  __bf16* wkT  = (__bf16*)alloc((size_t)DM * KV * HD * 2);
  __bf16* wvT  = (__bf16*)alloc((size_t)DM * KV * HD * 2);
  __bf16* woT  = (__bf16*)alloc((size_t)DM * DM * 2);
  float*  qf   = (float*)alloc(nBSD * 4);
  float*  kf   = (float*)alloc(nKVp * 4);
  float*  vf   = (float*)alloc(nKVp * 4);
  __bf16* qb   = (__bf16*)alloc(nBSD * 2);
  __bf16* kb   = (__bf16*)alloc(nKVp * 2);
  __bf16* vT   = (__bf16*)alloc(nKVp * 2);
  __bf16* ctx  = (__bf16*)alloc(nBSD * 2);

  // 1. x -> bf16
  cvt_bf16_kernel<<<(unsigned)((nBSD + 255) / 256), 256, 0, stream>>>(x, xb, nBSD);
  // 2. weight transposes (to [N,K] bf16)
  wtrans_kernel<<<dim3(DM / 32, DM / 32), dim3(32, 8), 0, stream>>>(wq, wqT, DM, DM);
  wtrans_kernel<<<dim3(KV * HD / 32, DM / 32), dim3(32, 8), 0, stream>>>(wk, wkT, DM, KV * HD);
  wtrans_kernel<<<dim3(KV * HD / 32, DM / 32), dim3(32, 8), 0, stream>>>(wv, wvT, DM, KV * HD);
  wtrans_kernel<<<dim3(DM / 32, DM / 32), dim3(32, 8), 0, stream>>>(wo, woT, H * HD, DM);
  // 3. QKV projections (async-LDS double-buffered WMMA GEMM)
  gemm_bf16_kernel<<<dim3(DM / 128, B * S / 128), 256, GEMM_LDS_BYTES, stream>>>(
      xb, wqT, qf, B * S, DM, DM);
  gemm_bf16_kernel<<<dim3(KV * HD / 128, B * S / 128), 256, GEMM_LDS_BYTES, stream>>>(
      xb, wkT, kf, B * S, KV * HD, DM);
  gemm_bf16_kernel<<<dim3(KV * HD / 128, B * S / 128), 256, GEMM_LDS_BYTES, stream>>>(
      xb, wvT, vf, B * S, KV * HD, DM);
  // 4. RoPE (+1/sqrt(HD) folded into q) and V transpose
  const float qscale = 0.08838834764831845f;  // 1/sqrt(128)
  size_t nq2 = (size_t)B * S * H * (HD / 2);
  size_t nk2 = (size_t)B * S * KV * (HD / 2);
  rope_kernel<<<(unsigned)((nq2 + 255) / 256), 256, 0, stream>>>(qf, fcos, fsin, qb, H, qscale);
  rope_kernel<<<(unsigned)((nk2 + 255) / 256), 256, 0, stream>>>(kf, fcos, fsin, kb, KV, 1.0f);
  vtrans_kernel<<<dim3(HD / 32, S / 32, B * KV), dim3(32, 8), 0, stream>>>(vf, vT);
  // 5. attention
  attn_kernel<<<dim3(S / 128, H, B), 256, 0, stream>>>(qb, kb, vT, ctx);
  // 6. output projection -> fp32 d_out
  gemm_bf16_kernel<<<dim3(DM / 128, B * S / 128), 256, GEMM_LDS_BYTES, stream>>>(
      ctx, woT, out, B * S, DM, DM);
}
